// MODEL_816043786789
// MI455X (gfx1250) — compile-verified, matrix operand order checked by reading, not run on hardware
//
#include <hip/hip_runtime.h>
#include <hip/hip_bf16.h>
#include <cstdint>

#define USER_NUM 100000
#define ITEM_NUM 50000
#define INTER    5
#define HIDE     64
#define MAX_LEN  240
#define N_BI     (USER_NUM + ITEM_NUM * INTER)   // 350000
#define E_BI     2000000
#define E_META   1000000
#define IN_SIZE  192

typedef __attribute__((ext_vector_type(16))) _Float16 v16h;
typedef __attribute__((ext_vector_type(8)))  _Float16 v8h;
typedef __attribute__((ext_vector_type(8)))  float    v8f;

// ---------------------------------------------------------------- utilities

__global__ void k_zero(float* __restrict__ p, long long n) {
  long long i = (long long)blockIdx.x * blockDim.x + threadIdx.x;
  long long stride = (long long)gridDim.x * blockDim.x;
  for (; i < n; i += stride) p[i] = 0.0f;
}

// proj[row][c] = bias[c] + sum_k tab(row,k) * W[k][c]; tab rows 0,1 forced to 0.
// tab(row,2i) = sin(row*div_i)/8, tab(row,2i+1)=cos(row*div_i)/8, div_i = 128/10000^(2i)
// (10000^(2i) overflows fp32 -> inf -> div 0, faithful to the reference)
__global__ void k_proj_table(const float* __restrict__ W, const float* __restrict__ bias,
                             float* __restrict__ out) {
  int row = blockIdx.x;   // 0..MAX_LEN-1
  int c   = threadIdx.x;  // 0..63
  float acc = bias[c];
  if (row >= 2) {
    for (int k = 0; k < 2 * HIDE; ++k) {
      int i = k >> 1;
      float denom = powf(10000.0f, (float)(2 * i));  // inf for large i (fp32 overflow)
      float div = 128.0f / denom;                    // 0 on overflow
      float ang = (float)row * div;
      float tval = (k & 1) ? cosf(ang) : sinf(ang);
      acc += tval * 0.125f * W[k * HIDE + c];
    }
  }
  out[row * HIDE + c] = acc;
}

__global__ void k_deg_count(const int* __restrict__ idx, float* __restrict__ deg, int E) {
  int i = blockIdx.x * blockDim.x + threadIdx.x;
  int stride = gridDim.x * blockDim.x;
  for (; i < E; i += stride) atomicAdd(&deg[idx[i]], 1.0f);
}

__global__ void k_deg_finish(float* __restrict__ deg, long long n) {
  long long i = (long long)blockIdx.x * blockDim.x + threadIdx.x;
  long long stride = (long long)gridDim.x * blockDim.x;
  for (; i < n; i += stride) {
    float d = deg[i];
    if (d < 1.0f) d = 1.0f;
    deg[i] = 1.0f / sqrtf(d);
  }
}

// agg[dst] += x[src] * inv_src[src]   (per feature)
__global__ void k_scatter(const float* __restrict__ x, const float* __restrict__ inv_src,
                          const int* __restrict__ src, const int* __restrict__ dst,
                          float* __restrict__ agg, long long total) {
  long long i = (long long)blockIdx.x * blockDim.x + threadIdx.x;
  long long stride = (long long)gridDim.x * blockDim.x;
  for (; i < total; i += stride) {
    long long e = i >> 6;
    int f = (int)(i & 63);
    int s = src[e];
    atomicAdd(&agg[(long long)dst[e] * HIDE + f], x[(long long)s * HIDE + f] * inv_src[s]);
  }
}

// rst[dst] += nf[src]*inv_src[src] + proj[tseq[e]]   (per feature)
__global__ void k_scatter_bi(const float* __restrict__ nf, const float* __restrict__ inv_src,
                             const int* __restrict__ src, const int* __restrict__ dst,
                             const int* __restrict__ tseq, const float* __restrict__ proj,
                             float* __restrict__ rst, long long total) {
  long long i = (long long)blockIdx.x * blockDim.x + threadIdx.x;
  long long stride = (long long)gridDim.x * blockDim.x;
  for (; i < total; i += stride) {
    long long e = i >> 6;
    int f = (int)(i & 63);
    int s = src[e];
    float v = nf[(long long)s * HIDE + f] * inv_src[s] + proj[tseq[e] * HIDE + f];
    atomicAdd(&rst[(long long)dst[e] * HIDE + f], v);
  }
}

// x[n][f] = leaky(x[n][f] * scale[n])
__global__ void k_rows_scale_leaky(float* __restrict__ x, const float* __restrict__ scale,
                                   long long total) {
  long long i = (long long)blockIdx.x * blockDim.x + threadIdx.x;
  long long stride = (long long)gridDim.x * blockDim.x;
  for (; i < total; i += stride) {
    long long n = i >> 6;
    float v = x[i] * scale[n];
    x[i] = v >= 0.0f ? v : 0.2f * v;
  }
}

__global__ void k_l2(const float* __restrict__ x, float* __restrict__ y, int N) {
  int n = blockIdx.x * blockDim.x + threadIdx.x;
  if (n >= N) return;
  long long b = (long long)n * HIDE;
  float s = 0.0f;
  for (int f = 0; f < HIDE; ++f) { float v = x[b + f]; s += v * v; }
  float nm = sqrtf(s);
  if (nm < 1e-12f) nm = 1e-12f;
  float inv = 1.0f / nm;
  for (int f = 0; f < HIDE; ++f) y[b + f] = x[b + f] * inv;
}

// --------------------------------------------------- WMMA GEMM  Y = act((rs*X) @ W)
// X:[N,64] W:[64,64] Y:[N,64].
// Block = 256 threads = 8 waves; W is staged transposed (f16) in LDS once per block.
// Each wave computes one 16-row tile x all 64 cols: 4 accumulators, 8 WMMAs,
// A fragment loaded once per K-chunk via 4x global_load_b128 and reused for 4 B tiles.
// Fragment layout per CDNA5 ISA 7.12.2: lane half=L>>4, mrow=L&15; element i of v16h
// maps to k = kc*32 + (i>>3)*16 + half*8 + (i&7)  -> two contiguous 8-element runs.
// act: 0 none, 1 prelu(*aptr), 2 leaky(0.2)
__global__ __launch_bounds__(256)
void k_gemm64(const float* __restrict__ X, const float* __restrict__ W,
              float* __restrict__ Y, int N,
              const float* __restrict__ rowscale,
              int act, const float* __restrict__ aptr) {
  __shared__ _Float16 Wt[HIDE * HIDE];  // transposed: Wt[col*64 + k]
  for (int t = threadIdx.x; t < HIDE * HIDE; t += 256) {
    int k = t >> 6, c = t & 63;
    Wt[c * HIDE + k] = (_Float16)W[t];
  }
  __syncthreads();

  int wave   = threadIdx.x >> 5;
  int tile_m = blockIdx.x * 8 + wave;
  if (tile_m * 16 >= N) return;            // wave-uniform exit (no barriers below)
  int lane = threadIdx.x & 31;
  int half = lane >> 4;
  int mrow = lane & 15;
  int grow = tile_m * 16 + mrow;
  bool rvalid = grow < N;
  int arow = rvalid ? grow : 0;
  float rs = rowscale ? rowscale[arow] : 1.0f;

  v8f acc[4] = {};
  for (int kc = 0; kc < 2; ++kc) {
    v16h a = {};
    if (rvalid) {
      const float* xr = X + (long long)arow * HIDE + kc * 32 + half * 8;
      float4 p0 = *(const float4*)(xr + 0);
      float4 p1 = *(const float4*)(xr + 4);
      float4 q0 = *(const float4*)(xr + 16);
      float4 q1 = *(const float4*)(xr + 20);
      a[0]  = (_Float16)(p0.x * rs); a[1]  = (_Float16)(p0.y * rs);
      a[2]  = (_Float16)(p0.z * rs); a[3]  = (_Float16)(p0.w * rs);
      a[4]  = (_Float16)(p1.x * rs); a[5]  = (_Float16)(p1.y * rs);
      a[6]  = (_Float16)(p1.z * rs); a[7]  = (_Float16)(p1.w * rs);
      a[8]  = (_Float16)(q0.x * rs); a[9]  = (_Float16)(q0.y * rs);
      a[10] = (_Float16)(q0.z * rs); a[11] = (_Float16)(q0.w * rs);
      a[12] = (_Float16)(q1.x * rs); a[13] = (_Float16)(q1.y * rs);
      a[14] = (_Float16)(q1.z * rs); a[15] = (_Float16)(q1.w * rs);
    }
#pragma unroll
    for (int tn = 0; tn < 4; ++tn) {
      const _Float16* bp = &Wt[(tn * 16 + mrow) * HIDE + kc * 32 + half * 8];
      v8h blo = *(const v8h*)(bp + 0);
      v8h bhi = *(const v8h*)(bp + 16);
      v16h b = __builtin_shufflevector(blo, bhi, 0, 1, 2, 3, 4, 5, 6, 7,
                                       8, 9, 10, 11, 12, 13, 14, 15);
      acc[tn] = __builtin_amdgcn_wmma_f32_16x16x32_f16(false, a, false, b,
                                                       (short)0, acc[tn], false, false);
    }
  }
  float alpha = (act == 1) ? aptr[0] : 0.2f;
#pragma unroll
  for (int tn = 0; tn < 4; ++tn) {
#pragma unroll
    for (int r = 0; r < 8; ++r) {
      int row = tile_m * 16 + r + half * 8;
      if (row < N) {
        float v = acc[tn][r];
        if (act != 0) v = (v >= 0.0f) ? v : alpha * v;
        Y[(long long)row * HIDE + tn * 16 + mrow] = v;
      }
    }
  }
}

// ------------------------------------------------- item fuse: mean over INTER copies
__global__ void k_ret_item(const float* __restrict__ item_emb, const float* __restrict__ ne0,
                           const float* __restrict__ ne1, float* __restrict__ out) {
  long long total = (long long)ITEM_NUM * IN_SIZE;
  long long i = (long long)blockIdx.x * blockDim.x + threadIdx.x;
  long long stride = (long long)gridDim.x * blockDim.x;
  for (; i < total; i += stride) {
    long long n = i / IN_SIZE;
    int d = (int)(i % IN_SIZE);
    int s = d >> 6, dd = d & 63;
    float acc = 0.0f;
    for (int r = 0; r < INTER; ++r) {
      long long row = n * INTER + r;
      float v;
      if (s == 0)      v = item_emb[row * HIDE + dd];
      else if (s == 1) v = ne0[(USER_NUM + row) * HIDE + dd];
      else             v = ne1[(USER_NUM + row) * HIDE + dd];
      acc += v;
    }
    out[i] = acc * (1.0f / INTER);
  }
}

// ------------------------------------------------- semantic attention scoring (WMMA)
// partial += sum over nodes of  tanh(z @ W1 + b1) @ W2, where
// z[n][k] = segK[k>>6][n*rowStride + (k&63)]  (three 64-wide segments).
// Block = 256 threads = 8 waves; W1 (192x128) staged transposed f16 in LDS (48 KB).
// Each wave: 16-node tile x 128 cols = 8 accumulators; K=192 -> 6 chunks -> 48 WMMAs.
__global__ __launch_bounds__(256)
void k_att_wmma(const float* __restrict__ s0, const float* __restrict__ s1,
                const float* __restrict__ s2, int rowStride,
                const float* __restrict__ W1, const float* __restrict__ b1,
                const float* __restrict__ W2,
                int N, float* __restrict__ partial) {
  __shared__ _Float16 W1t[128 * IN_SIZE];  // W1t[col*192 + k]
  for (int t = threadIdx.x; t < 128 * IN_SIZE; t += 256) {
    int k = t >> 7, c = t & 127;
    W1t[c * IN_SIZE + k] = (_Float16)W1[t];
  }
  __syncthreads();

  int wave   = threadIdx.x >> 5;
  int tile_m = blockIdx.x * 8 + wave;
  int lane = threadIdx.x & 31;
  float lsum = 0.0f;
  if (tile_m * 16 < N) {                       // wave-uniform
    int half = lane >> 4;
    int mrow = lane & 15;
    int grow = tile_m * 16 + mrow;
    bool rvalid = grow < N;
    int arow = rvalid ? grow : 0;

    v8f acc[8] = {};
    for (int kc = 0; kc < 6; ++kc) {
      v16h a = {};
      if (rvalid) {
        int k1 = kc * 32 + half * 8;           // first 8-run
        int k2 = k1 + 16;                      // second 8-run
        const float* sp1 = (k1 < 64 ? s0 : k1 < 128 ? s1 : s2);
        const float* sp2 = (k2 < 64 ? s0 : k2 < 128 ? s1 : s2);
        const float* r1 = sp1 + (long long)arow * rowStride + (k1 & 63);
        const float* r2 = sp2 + (long long)arow * rowStride + (k2 & 63);
        float4 p0 = *(const float4*)(r1 + 0);
        float4 p1 = *(const float4*)(r1 + 4);
        float4 q0 = *(const float4*)(r2 + 0);
        float4 q1 = *(const float4*)(r2 + 4);
        a[0]  = (_Float16)p0.x; a[1]  = (_Float16)p0.y;
        a[2]  = (_Float16)p0.z; a[3]  = (_Float16)p0.w;
        a[4]  = (_Float16)p1.x; a[5]  = (_Float16)p1.y;
        a[6]  = (_Float16)p1.z; a[7]  = (_Float16)p1.w;
        a[8]  = (_Float16)q0.x; a[9]  = (_Float16)q0.y;
        a[10] = (_Float16)q0.z; a[11] = (_Float16)q0.w;
        a[12] = (_Float16)q1.x; a[13] = (_Float16)q1.y;
        a[14] = (_Float16)q1.z; a[15] = (_Float16)q1.w;
      }
#pragma unroll
      for (int tn = 0; tn < 8; ++tn) {
        const _Float16* bp = &W1t[(tn * 16 + mrow) * IN_SIZE + kc * 32 + half * 8];
        v8h blo = *(const v8h*)(bp + 0);
        v8h bhi = *(const v8h*)(bp + 16);
        v16h b = __builtin_shufflevector(blo, bhi, 0, 1, 2, 3, 4, 5, 6, 7,
                                         8, 9, 10, 11, 12, 13, 14, 15);
        acc[tn] = __builtin_amdgcn_wmma_f32_16x16x32_f16(false, a, false, b,
                                                         (short)0, acc[tn], false, false);
      }
    }
    // tanh-weighted reduction: lane owns cols {tn*16+mrow}, rows {tile_m*16 + r + 8*half}
#pragma unroll
    for (int tn = 0; tn < 8; ++tn) {
      int col = tn * 16 + mrow;
      float bb = b1[col], w2 = W2[col];
#pragma unroll
      for (int r = 0; r < 8; ++r) {
        int row = tile_m * 16 + r + half * 8;
        if (row < N) lsum += tanhf(acc[tn][r] + bb) * w2;
      }
    }
  }
  // wave32 reduction, one atomic per wave
  for (int o = 16; o >= 1; o >>= 1) lsum += __shfl_xor(lsum, o, 32);
  if (lane == 0 && tile_m * 16 < N) atomicAdd(partial, lsum);
}

__global__ void k_softmax3(const float* __restrict__ partials, float invN,
                           float* __restrict__ beta) {
  if (blockIdx.x == 0 && threadIdx.x == 0) {
    float w0 = partials[0] * invN, w1 = partials[1] * invN, w2 = partials[2] * invN;
    float m = fmaxf(w0, fmaxf(w1, w2));
    float e0 = expf(w0 - m), e1 = expf(w1 - m), e2 = expf(w2 - m);
    float s = e0 + e1 + e2;
    beta[0] = e0 / s; beta[1] = e1 / s; beta[2] = e2 / s;
  }
}

// out[n][d] = sum_p beta[p] * z_p[n][d]
__global__ void k_mix(const float* __restrict__ base,
                      const float* __restrict__ p0s1, const float* __restrict__ p0s2,
                      const float* __restrict__ p1s1, const float* __restrict__ p1s2,
                      const float* __restrict__ p2s1, const float* __restrict__ p2s2,
                      const float* __restrict__ p2wide,
                      const float* __restrict__ beta, float* __restrict__ out, int N) {
  long long total = (long long)N * IN_SIZE;
  long long i = (long long)blockIdx.x * blockDim.x + threadIdx.x;
  long long stride = (long long)gridDim.x * blockDim.x;
  float b0 = beta[0], b1v = beta[1], b2 = beta[2];
  for (; i < total; i += stride) {
    long long n = i / IN_SIZE;
    int d = (int)(i % IN_SIZE);
    int s = d >> 6, dd = d & 63;
    long long o = n * HIDE + dd;
    float z0 = (s == 0) ? base[o] : (s == 1) ? p0s1[o] : p0s2[o];
    float z1 = (s == 0) ? base[o] : (s == 1) ? p1s1[o] : p1s2[o];
    float z2;
    if (p2wide) z2 = p2wide[i];  // i == n*IN_SIZE + d
    else        z2 = (s == 0) ? base[o] : (s == 1) ? p2s1[o] : p2s2[o];
    out[i] = b0 * z0 + b1v * z1 + b2 * z2;
  }
}

// ---------------------------------------------------------------- host driver

extern "C" void kernel_launch(void* const* d_in, const int* in_sizes, int n_in,
                              void* d_out, int out_size, void* d_ws, size_t ws_size,
                              hipStream_t stream) {
  (void)in_sizes; (void)n_in; (void)out_size; (void)ws_size;
  const float* user_emb  = (const float*)d_in[0];
  const float* item_emb  = (const float*)d_in[1];
  const float* item_emb2 = (const float*)d_in[2];
  const float* te_W      = (const float*)d_in[3];
  const float* te_b      = (const float*)d_in[4];
  const float* gcn0_uw   = (const float*)d_in[5];
  const float* gcn0_vw   = (const float*)d_in[6];
  const float* gcn1_uw   = (const float*)d_in[7];
  const float* gcn1_vw   = (const float*)d_in[8];
  const float* umeta_w   = (const float*)d_in[9];
  const float* imeta_w   = (const float*)d_in[10];
  const float* prelu_a   = (const float*)d_in[11];
  const float* saU_W1    = (const float*)d_in[12];
  const float* saU_b1    = (const float*)d_in[13];
  const float* saU_W2    = (const float*)d_in[14];
  const float* saI_W1    = (const float*)d_in[15];
  const float* saI_b1    = (const float*)d_in[16];
  const float* saI_W2    = (const float*)d_in[17];
  const int*   bi_src    = (const int*)d_in[18];
  const int*   bi_dst    = (const int*)d_in[19];
  const int*   tseq      = (const int*)d_in[20];
  const int*   ug_src    = (const int*)d_in[21];
  const int*   ug_dst    = (const int*)d_in[22];
  const int*   ig_src    = (const int*)d_in[23];
  const int*   ig_dst    = (const int*)d_in[24];
  float* out = (float*)d_out;

  // workspace carve-out
  float* w = (float*)d_ws;
  size_t off = 0;
  auto alloc = [&](size_t nf_) { float* p = w + off; off += nf_; return p; };
  float* proj   = alloc((size_t)MAX_LEN * HIDE);
  float* inv_bs = alloc(N_BI);
  float* inv_bd = alloc(N_BI);
  float* invUs[2], *invUd[2], *invIs[2], *invId[2];
  for (int p = 0; p < 2; ++p) { invUs[p] = alloc(USER_NUM); invUd[p] = alloc(USER_NUM); }
  for (int p = 0; p < 2; ++p) { invIs[p] = alloc(ITEM_NUM); invId[p] = alloc(ITEM_NUM); }
  long long degTotal = 2LL * N_BI + 4LL * USER_NUM + 4LL * ITEM_NUM;
  size_t nbif = (size_t)N_BI * HIDE;
  float* nf    = alloc(nbif);
  float* emb0  = alloc(nbif);
  float* ne0   = alloc(nbif);
  float* emb1  = alloc(nbif);
  float* ne1   = alloc(nbif);
  float* xu    = alloc((size_t)USER_NUM * HIDE);
  float* aggU  = alloc((size_t)USER_NUM * HIDE);
  float* ul2[2][2];
  for (int p = 0; p < 2; ++p) for (int j = 0; j < 2; ++j) ul2[p][j] = alloc((size_t)USER_NUM * HIDE);
  float* xi    = alloc((size_t)ITEM_NUM * HIDE);
  float* aggI  = alloc((size_t)ITEM_NUM * HIDE);
  float* il2[2][2];
  for (int p = 0; p < 2; ++p) for (int j = 0; j < 2; ++j) il2[p][j] = alloc((size_t)ITEM_NUM * HIDE);
  float* ret_i = alloc((size_t)ITEM_NUM * IN_SIZE);
  float* partU = alloc(4);
  float* partI = alloc(4);
  float* betaU = alloc(4);
  float* betaI = alloc(4);

  const dim3 T256(256);
  const dim3 GS(4096);  // grid-stride kernels

  // 1) time-encoding projection table [240,64]
  k_proj_table<<<MAX_LEN, HIDE, 0, stream>>>(te_W, te_b, proj);

  // 2) degrees -> inv sqrt
  k_zero<<<GS, T256, 0, stream>>>(inv_bs, degTotal);  // deg arrays are contiguous
  k_deg_count<<<GS, T256, 0, stream>>>(bi_src, inv_bs, E_BI);
  k_deg_count<<<GS, T256, 0, stream>>>(bi_dst, inv_bd, E_BI);
  for (int p = 0; p < 2; ++p) {
    k_deg_count<<<GS, T256, 0, stream>>>(ug_src + (size_t)p * E_META, invUs[p], E_META);
    k_deg_count<<<GS, T256, 0, stream>>>(ug_dst + (size_t)p * E_META, invUd[p], E_META);
    k_deg_count<<<GS, T256, 0, stream>>>(ig_src + (size_t)p * E_META, invIs[p], E_META);
    k_deg_count<<<GS, T256, 0, stream>>>(ig_dst + (size_t)p * E_META, invId[p], E_META);
  }
  k_deg_finish<<<GS, T256, 0, stream>>>(inv_bs, degTotal);

  const int ITEMROWS = ITEM_NUM * INTER;
  auto gw = [](int n) {  // blocks of 8 row-tiles (16 rows each)
    return dim3((unsigned)(((n + 15) / 16 + 7) / 8));
  };

  // 3) bipartite GCN layer 0: nf = [u@uw ; v@vw], scatter(+edge feat), dst-norm, leaky
  k_gemm64<<<gw(USER_NUM), T256, 0, stream>>>(user_emb, gcn0_uw, nf, USER_NUM, nullptr, 0, nullptr);
  k_gemm64<<<gw(ITEMROWS), T256, 0, stream>>>(item_emb, gcn0_vw, nf + (size_t)USER_NUM * HIDE,
                                              ITEMROWS, nullptr, 0, nullptr);
  k_zero<<<GS, T256, 0, stream>>>(emb0, (long long)nbif);
  k_scatter_bi<<<GS, T256, 0, stream>>>(nf, inv_bs, bi_src, bi_dst, tseq, proj, emb0,
                                        (long long)E_BI * HIDE);
  k_rows_scale_leaky<<<GS, T256, 0, stream>>>(emb0, inv_bd, (long long)nbif);
  k_l2<<<(N_BI + 255) / 256, T256, 0, stream>>>(emb0, ne0, N_BI);

  // 4) bipartite GCN layer 1
  k_gemm64<<<gw(USER_NUM), T256, 0, stream>>>(emb0, gcn1_uw, nf, USER_NUM, nullptr, 0, nullptr);
  k_gemm64<<<gw(ITEMROWS), T256, 0, stream>>>(emb0 + (size_t)USER_NUM * HIDE, gcn1_vw,
                                              nf + (size_t)USER_NUM * HIDE, ITEMROWS,
                                              nullptr, 0, nullptr);
  k_zero<<<GS, T256, 0, stream>>>(emb1, (long long)nbif);
  k_scatter_bi<<<GS, T256, 0, stream>>>(nf, inv_bs, bi_src, bi_dst, tseq, proj, emb1,
                                        (long long)E_BI * HIDE);
  k_rows_scale_leaky<<<GS, T256, 0, stream>>>(emb1, inv_bd, (long long)nbif);
  k_l2<<<(N_BI + 255) / 256, T256, 0, stream>>>(emb1, ne1, N_BI);

  // 5) user meta-path convs: agg = dstnorm * segsum(srcnorm * x); y = prelu(agg @ W); store l2(y)
  for (int p = 0; p < 2; ++p) {
    const int* us = ug_src + (size_t)p * E_META;
    const int* ud = ug_dst + (size_t)p * E_META;
    const float* xin = user_emb;
    for (int j = 0; j < 2; ++j) {
      k_zero<<<GS, T256, 0, stream>>>(aggU, (long long)USER_NUM * HIDE);
      k_scatter<<<GS, T256, 0, stream>>>(xin, invUs[p], us, ud, aggU, (long long)E_META * HIDE);
      k_gemm64<<<gw(USER_NUM), T256, 0, stream>>>(aggU, umeta_w + ((size_t)p * 2 + j) * HIDE * HIDE,
                                                  xu, USER_NUM, invUd[p], 1, prelu_a);
      k_l2<<<(USER_NUM + 255) / 256, T256, 0, stream>>>(xu, ul2[p][j], USER_NUM);
      xin = xu;
    }
  }

  // 6) item meta-path convs
  for (int p = 0; p < 2; ++p) {
    const int* is_ = ig_src + (size_t)p * E_META;
    const int* id_ = ig_dst + (size_t)p * E_META;
    const float* xin = item_emb2;
    for (int j = 0; j < 2; ++j) {
      k_zero<<<GS, T256, 0, stream>>>(aggI, (long long)ITEM_NUM * HIDE);
      k_scatter<<<GS, T256, 0, stream>>>(xin, invIs[p], is_, id_, aggI, (long long)E_META * HIDE);
      k_gemm64<<<gw(ITEM_NUM), T256, 0, stream>>>(aggI, imeta_w + ((size_t)p * 2 + j) * HIDE * HIDE,
                                                  xi, ITEM_NUM, invId[p], 1, prelu_a);
      k_l2<<<(ITEM_NUM + 255) / 256, T256, 0, stream>>>(xi, il2[p][j], ITEM_NUM);
      xin = xi;
    }
  }

  // 7) item 5-way fuse (mean)
  k_ret_item<<<GS, T256, 0, stream>>>(item_emb, ne0, ne1, ret_i);

  // 8) semantic attention (WMMA scoring)
  k_zero<<<1, T256, 0, stream>>>(partU, 8);  // partU + partI contiguous
  // users: z_p segments, all stride 64
  k_att_wmma<<<gw(USER_NUM), T256, 0, stream>>>(user_emb, ul2[0][0], ul2[0][1], HIDE,
                                                saU_W1, saU_b1, saU_W2, USER_NUM, partU + 0);
  k_att_wmma<<<gw(USER_NUM), T256, 0, stream>>>(user_emb, ul2[1][0], ul2[1][1], HIDE,
                                                saU_W1, saU_b1, saU_W2, USER_NUM, partU + 1);
  k_att_wmma<<<gw(USER_NUM), T256, 0, stream>>>(user_emb, ne0, ne1, HIDE,
                                                saU_W1, saU_b1, saU_W2, USER_NUM, partU + 2);
  // items: p2 is the contiguous [N,192] fused tensor -> segments with stride 192
  k_att_wmma<<<gw(ITEM_NUM), T256, 0, stream>>>(item_emb2, il2[0][0], il2[0][1], HIDE,
                                                saI_W1, saI_b1, saI_W2, ITEM_NUM, partI + 0);
  k_att_wmma<<<gw(ITEM_NUM), T256, 0, stream>>>(item_emb2, il2[1][0], il2[1][1], HIDE,
                                                saI_W1, saI_b1, saI_W2, ITEM_NUM, partI + 1);
  k_att_wmma<<<gw(ITEM_NUM), T256, 0, stream>>>(ret_i, ret_i + HIDE, ret_i + 2 * HIDE, IN_SIZE,
                                                saI_W1, saI_b1, saI_W2, ITEM_NUM, partI + 2);
  k_softmax3<<<1, 1, 0, stream>>>(partU, 1.0f / (float)USER_NUM, betaU);
  k_softmax3<<<1, 1, 0, stream>>>(partI, 1.0f / (float)ITEM_NUM, betaI);

  // 9) weighted mix into outputs
  k_mix<<<GS, T256, 0, stream>>>(user_emb, ul2[0][0], ul2[0][1], ul2[1][0], ul2[1][1],
                                 ne0, ne1, nullptr, betaU, out, USER_NUM);
  k_mix<<<GS, T256, 0, stream>>>(item_emb2, il2[0][0], il2[0][1], il2[1][0], il2[1][1],
                                 nullptr, nullptr, ret_i, betaI,
                                 out + (size_t)USER_NUM * IN_SIZE, ITEM_NUM);
}
